// Attention_57380763075267
// MI455X (gfx1250) — matrix operand order repaired and checked by reading, not compiled, not run on hardware
//
#include <hip/hip_runtime.h>

typedef __attribute__((ext_vector_type(16))) _Float16 v16h;
typedef __attribute__((ext_vector_type(8)))  _Float16 v8h;
typedef __attribute__((ext_vector_type(8)))  float    v8f;
typedef __attribute__((ext_vector_type(4)))  unsigned int u32x4;
typedef __attribute__((ext_vector_type(8)))  int          i32x8;
typedef __attribute__((ext_vector_type(4)))  int          i32x4;

#define B_   4
#define N_   2048
#define DIM_ 512
#define H_   8
#define HD_  64
#define SCALE_ 0.125f   // 1/sqrt(64)
#define NEG_BIG -3.0e38f

#if defined(__has_builtin)
# if __has_builtin(__builtin_amdgcn_tensor_load_to_lds)
#  define HAVE_TDM 1
# endif
#endif
#if defined(__has_include)
# if __has_include(<hip/amd_detail/amd_gfx1250_TDM.h>)
#  define TDM_SIX_ARGS 1   // therock-10.0 headers => clang-23 6-arg builtin
# endif
#endif

union AFrag { v16h v; v8h h[2]; };

// ---------------------------------------------------------------- converts
__global__ void cvt_f32_f16(const float* __restrict__ in, _Float16* __restrict__ out, int n) {
    int i = blockIdx.x * blockDim.x + threadIdx.x;
    if (i < n) out[i] = (_Float16)in[i];
}

// ---------------------------------------------------------------- QKV GEMM
// qkv = x @ qkv_w.T ; cols 0..511 -> Q [B,H,N,HD], cols 512..1023 -> V^T [B,H,HD,N]
// 16x64 output per wave: one A fragment reused across 4 B tiles.
__global__ __launch_bounds__(256) void qkv_gemm(const _Float16* __restrict__ x16,
                                                const _Float16* __restrict__ w1,
                                                _Float16* __restrict__ q16,
                                                _Float16* __restrict__ vt16) {
    const int lane = threadIdx.x & 31;
    const int wave = threadIdx.x >> 5;
    const int lo   = lane & 15;
    const int hsel = lane >> 4;
    const int hi8  = hsel * 8;

    const int    ncol0 = blockIdx.x * 64;                      // 0..960
    const size_t mrow  = ((size_t)blockIdx.y * 8 + wave) * 16; // 0..8176

    const _Float16* arow = x16 + (mrow + lo) * DIM_;
    const _Float16* br[4];
#pragma unroll
    for (int c = 0; c < 4; ++c)
        br[c] = w1 + (size_t)(ncol0 + c * 16 + lo) * DIM_ + hsel * 16;

    v8f acc[4] = {};
    for (int k0 = 0; k0 < DIM_; k0 += 32) {
        AFrag a;
        a.h[0] = *(const v8h*)(arow + k0 + hsel * 16);
        a.h[1] = *(const v8h*)(arow + k0 + hsel * 16 + 8);
#pragma unroll
        for (int c = 0; c < 4; ++c) {
            v16h b = *(const v16h*)(br[c] + k0);
            acc[c] = __builtin_amdgcn_wmma_f32_16x16x32_f16(false, a.v, false, b,
                                                            (short)0, acc[c], false, false);
        }
    }

#pragma unroll
    for (int c = 0; c < 4; ++c) {
        const int n = ncol0 + c * 16 + lo;
#pragma unroll
        for (int i = 0; i < 8; ++i) {
            size_t m   = mrow + i + hsel * 8;
            int    bi  = (int)(m / N_);
            int    seq = (int)(m % N_);
            _Float16 val = (_Float16)acc[c][i];
            if (n < DIM_) {                   // Q: [B,H,N,HD]
                int h = n / HD_, d = n % HD_;
                q16[(((size_t)bi * H_ + h) * N_ + seq) * HD_ + d] = val;
            } else {                          // V transposed: [B,H,HD,N]
                int nn = n - DIM_;
                int h = nn / HD_, d = nn % HD_;
                vt16[(((size_t)bi * H_ + h) * HD_ + d) * N_ + seq] = val;
            }
        }
    }
}

// ---------------------------------------------------------------- attention
// per (b,h): S = Q K^T * scale (k == q), mask -> -big, online softmax, O = P V
__global__ __launch_bounds__(256) void attn_kernel(const _Float16* __restrict__ q16,
                                                   const _Float16* __restrict__ vt16,
                                                   const int* __restrict__ mask,
                                                   _Float16* __restrict__ o16) {
    __shared__ __align__(64) _Float16 lds_p[8][16 * 32];   // per-wave P staging
#ifdef HAVE_TDM
    __shared__ __align__(128) int smask[128 * 32];         // TDM-fed mask tile
#endif

    const int lane = threadIdx.x & 31;
    const int wave = threadIdx.x >> 5;
    const int lo   = lane & 15;
    const int hsel = lane >> 4;
    const int hi8  = hsel * 8;

    const int bh      = blockIdx.y;            // 0..31
    const int b       = bh / H_;
    const int h       = bh % H_;
    const int rowbase = blockIdx.x * 128;      // block's first query row
    const int m0      = rowbase + wave * 16;

    const _Float16* qb = q16  + (size_t)bh * N_ * HD_;
    const _Float16* vb = vt16 + (size_t)bh * HD_ * N_;
    _Float16* myp = lds_p[wave];

    // Q A-fragments for d-chunks [0,32) and [32,64)
    AFrag qa0, qa1;
    {
        const _Float16* qr = qb + (size_t)(m0 + lo) * HD_;
        qa0.h[0] = *(const v8h*)(qr + hsel * 16);
        qa0.h[1] = *(const v8h*)(qr + hsel * 16 + 8);
        qa1.h[0] = *(const v8h*)(qr + 32 + hsel * 16);
        qa1.h[1] = *(const v8h*)(qr + 32 + hsel * 16 + 8);
    }

    v8f o0 = {}, o1 = {}, o2 = {}, o3 = {};
    float rm[8], rl[8];
#pragma unroll
    for (int i = 0; i < 8; ++i) { rm[i] = NEG_BIG; rl[i] = 0.0f; }

    for (int j0 = 0; j0 < N_; j0 += 32) {
#ifdef HAVE_TDM
        // Tensor Data Mover: DMA the block's 128x32 int32 mask tile into LDS.
        if (wave == 0) {
            unsigned long long ga =
                (unsigned long long)(uintptr_t)(mask + (size_t)rowbase * N_ + j0);
            unsigned lds_addr =
                (unsigned)(unsigned long long)(uintptr_t)(void*)smask;
            u32x4 g0;
            g0[0] = 1u;                                        // count=1 (valid)
            g0[1] = lds_addr;                                  // lds_addr[31:0]
            g0[2] = (unsigned)(ga & 0xFFFFFFFFu);              // global_addr lo
            g0[3] = (unsigned)((ga >> 32) & 0x01FFFFFFu) | 0x80000000u; // hi + type=2
            i32x8 g1;
            g1[0] = 0x00020000;          // data_size = 4B, no multicast
            g1[1] = (int)(2048u << 16);  // tensor_dim0 = 2048 (lo16 in [31:16])
            g1[2] = (int)(2048u << 16);  // tensor_dim0 hi=0 | tensor_dim1 lo16
            g1[3] = (int)(32u << 16);    // tensor_dim1 hi=0 | tile_dim0 = 32
            g1[4] = 128;                 // tile_dim1 = 128, tile_dim2 = 0
            g1[5] = 2048;                // tensor_dim0_stride = 2048
            g1[6] = 0;                   // stride hi | tensor_dim1_stride lo
            g1[7] = 0;
            i32x4 g2 = {0, 0, 0, 0};
            i32x4 g3 = {0, 0, 0, 0};
#ifdef TDM_SIX_ARGS
            i32x8 g4 = {0, 0, 0, 0, 0, 0, 0, 0};
            __builtin_amdgcn_tensor_load_to_lds(g0, g1, g2, g3, g4, 0);
#else
            __builtin_amdgcn_tensor_load_to_lds(g0, g1, g2, g3, 0);
#endif
            __builtin_amdgcn_s_wait_tensorcnt(0);
        }
        __syncthreads();   // mask tile visible to all waves
#endif

        // K B-fragments: key tile 0 = rows j0..j0+15, tile 1 = j0+16..j0+31
        const _Float16* kr0 = qb + (size_t)(j0 + lo) * HD_;
        const _Float16* kr1 = qb + (size_t)(j0 + 16 + lo) * HD_;
        v16h k0a = *(const v16h*)(kr0 + hsel * 16);
        v16h k0b = *(const v16h*)(kr0 + 32 + hsel * 16);
        v16h k1a = *(const v16h*)(kr1 + hsel * 16);
        v16h k1b = *(const v16h*)(kr1 + 32 + hsel * 16);

        v8f s0 = {}, s1 = {};
        s0 = __builtin_amdgcn_wmma_f32_16x16x32_f16(false, qa0.v, false, k0a, (short)0, s0, false, false);
        s0 = __builtin_amdgcn_wmma_f32_16x16x32_f16(false, qa1.v, false, k0b, (short)0, s0, false, false);
        s1 = __builtin_amdgcn_wmma_f32_16x16x32_f16(false, qa0.v, false, k1a, (short)0, s1, false, false);
        s1 = __builtin_amdgcn_wmma_f32_16x16x32_f16(false, qa1.v, false, k1b, (short)0, s1, false, false);

        float corr[8];
#pragma unroll
        for (int i = 0; i < 8; ++i) {
            const int rloc = wave * 16 + i + hsel * 8;   // row within block tile
#ifdef HAVE_TDM
            const int c0 = smask[rloc * 32 + lo];
            const int c1 = smask[rloc * 32 + 16 + lo];
#else
            const int mrow = rowbase + rloc;
            const int c0 = mask[(size_t)mrow * N_ + j0 + lo];
            const int c1 = mask[(size_t)mrow * N_ + j0 + 16 + lo];
#endif
            float a = c0 ? NEG_BIG : s0[i] * SCALE_;
            float c = c1 ? NEG_BIG : s1[i] * SCALE_;
            float t = fmaxf(a, c);
            t = fmaxf(t, __shfl_xor(t, 1, 32));
            t = fmaxf(t, __shfl_xor(t, 2, 32));
            t = fmaxf(t, __shfl_xor(t, 4, 32));
            t = fmaxf(t, __shfl_xor(t, 8, 32));
            float mn = fmaxf(rm[i], t);
            float e0 = __expf(a - mn);
            float e1 = __expf(c - mn);
            float ts = e0 + e1;
            ts += __shfl_xor(ts, 1, 32);
            ts += __shfl_xor(ts, 2, 32);
            ts += __shfl_xor(ts, 4, 32);
            ts += __shfl_xor(ts, 8, 32);
            float cr = __expf(rm[i] - mn);
            rl[i] = rl[i] * cr + ts;
            rm[i] = mn;
            corr[i] = cr;
            // stage P (16x32) row-major into this wave's LDS region
            myp[(i + hsel * 8) * 32 + lo]      = (_Float16)e0;
            myp[(i + hsel * 8) * 32 + 16 + lo] = (_Float16)e1;
        }
#pragma unroll
        for (int i = 0; i < 8; ++i) {
            o0[i] *= corr[i]; o1[i] *= corr[i]; o2[i] *= corr[i]; o3[i] *= corr[i];
        }

        // re-read P as A-fragment (wave-internal DS ops are in order; no barrier)
        AFrag pf;
        pf.h[0] = *(const v8h*)(myp + lo * 32 + hsel * 16);
        pf.h[1] = *(const v8h*)(myp + lo * 32 + hsel * 16 + 8);

        // V B-fragments: column d contiguous over k thanks to V^T layout
        v16h vf0 = *(const v16h*)(vb + (size_t)(0 * 16 + lo) * N_ + j0 + hsel * 16);
        v16h vf1 = *(const v16h*)(vb + (size_t)(1 * 16 + lo) * N_ + j0 + hsel * 16);
        v16h vf2 = *(const v16h*)(vb + (size_t)(2 * 16 + lo) * N_ + j0 + hsel * 16);
        v16h vf3 = *(const v16h*)(vb + (size_t)(3 * 16 + lo) * N_ + j0 + hsel * 16);

        o0 = __builtin_amdgcn_wmma_f32_16x16x32_f16(false, pf.v, false, vf0, (short)0, o0, false, false);
        o1 = __builtin_amdgcn_wmma_f32_16x16x32_f16(false, pf.v, false, vf1, (short)0, o1, false, false);
        o2 = __builtin_amdgcn_wmma_f32_16x16x32_f16(false, pf.v, false, vf2, (short)0, o2, false, false);
        o3 = __builtin_amdgcn_wmma_f32_16x16x32_f16(false, pf.v, false, vf3, (short)0, o3, false, false);

#ifdef HAVE_TDM
        __syncthreads();   // everyone done with smask before next DMA overwrites it
#endif
    }

    // normalize + write O as [B, N, DIM] f16 for the output projection
#pragma unroll
    for (int i = 0; i < 8; ++i) {
        float inv = 1.0f / rl[i];
        int mrow = m0 + i + hsel * 8;
        size_t rowoff = ((size_t)b * N_ + mrow) * DIM_ + (size_t)h * HD_;
        o16[rowoff + 0 * 16 + lo] = (_Float16)(o0[i] * inv);
        o16[rowoff + 1 * 16 + lo] = (_Float16)(o1[i] * inv);
        o16[rowoff + 2 * 16 + lo] = (_Float16)(o2[i] * inv);
        o16[rowoff + 3 * 16 + lo] = (_Float16)(o3[i] * inv);
    }
}

// ---------------------------------------------------------------- out proj
// 16x64 output per wave, 4 accumulators sharing one A fragment.
__global__ __launch_bounds__(256) void proj_gemm(const _Float16* __restrict__ o16,
                                                 const _Float16* __restrict__ w2,
                                                 const float* __restrict__ bias,
                                                 float* __restrict__ out) {
    const int lane = threadIdx.x & 31;
    const int wave = threadIdx.x >> 5;
    const int lo   = lane & 15;
    const int hsel = lane >> 4;
    const int hi8  = hsel * 8;

    const int    ncol0 = blockIdx.x * 64;                      // 0..448
    const size_t mrow  = ((size_t)blockIdx.y * 8 + wave) * 16;

    const _Float16* arow = o16 + (mrow + lo) * DIM_;
    const _Float16* br[4];
#pragma unroll
    for (int c = 0; c < 4; ++c)
        br[c] = w2 + (size_t)(ncol0 + c * 16 + lo) * DIM_ + hsel * 16;

    v8f acc[4] = {};
    for (int k0 = 0; k0 < DIM_; k0 += 32) {
        AFrag a;
        a.h[0] = *(const v8h*)(arow + k0 + hsel * 16);
        a.h[1] = *(const v8h*)(arow + k0 + hsel * 16 + 8);
#pragma unroll
        for (int c = 0; c < 4; ++c) {
            v16h b = *(const v16h*)(br[c] + k0);
            acc[c] = __builtin_amdgcn_wmma_f32_16x16x32_f16(false, a.v, false, b,
                                                            (short)0, acc[c], false, false);
        }
    }

#pragma unroll
    for (int c = 0; c < 4; ++c) {
        const int n = ncol0 + c * 16 + lo;
        float bb = bias[n];
#pragma unroll
        for (int i = 0; i < 8; ++i) {
            size_t m = mrow + i + hsel * 8;
            out[m * DIM_ + n] = acc[c][i] + bb;
        }
    }
}

// ---------------------------------------------------------------- launcher
extern "C" void kernel_launch(void* const* d_in, const int* in_sizes, int n_in,
                              void* d_out, int out_size, void* d_ws, size_t ws_size,
                              hipStream_t stream) {
    const float* x      = (const float*)d_in[0];
    const float* qkv_w  = (const float*)d_in[1];
    const float* proj_w = (const float*)d_in[2];
    const float* proj_b = (const float*)d_in[3];
    const int*   mask   = (const int*)d_in[4];
    float*       out    = (float*)d_out;

    const size_t nX  = (size_t)B_ * N_ * DIM_;   // 4,194,304
    const size_t nW1 = (size_t)2 * DIM_ * DIM_;  // 524,288
    const size_t nW2 = (size_t)DIM_ * DIM_;      // 262,144

    _Float16* x16  = (_Float16*)d_ws;
    _Float16* w1   = x16  + nX;
    _Float16* w2   = w1   + nW1;
    _Float16* q16  = w2   + nW2;
    _Float16* vt16 = q16  + nX;
    _Float16* o16  = vt16 + nX;

    cvt_f32_f16<<<(int)((nX  + 255) / 256), 256, 0, stream>>>(x,      x16, (int)nX);
    cvt_f32_f16<<<(int)((nW1 + 255) / 256), 256, 0, stream>>>(qkv_w,  w1,  (int)nW1);
    cvt_f32_f16<<<(int)((nW2 + 255) / 256), 256, 0, stream>>>(proj_w, w2,  (int)nW2);

    // QKV: M=8192 (128 rows/block), Nout=1024 (64 cols/block)
    qkv_gemm<<<dim3(16, 64), 256, 0, stream>>>(x16, w1, q16, vt16);

    // attention: 16 query-blocks of 128 rows x 32 (b,h) pairs
    attn_kernel<<<dim3(16, 32), 256, 0, stream>>>(q16, vt16, mask, o16);

    // projection: Nout=512 (64 cols/block), M=8192 (128 rows/block)
    proj_gemm<<<dim3(8, 64), 256, 0, stream>>>(o16, w2, proj_b, out);
}
